// MEGADecoder_35390530519890
// MI455X (gfx1250) — matrix operand order repaired and checked by reading, not compiled, not run on hardware
//
#include <hip/hip_runtime.h>
#include <hip/hip_bf16.h>
#include <math.h>

// ---------------- types / WMMA helpers ----------------
typedef __bf16 bf16;
typedef __attribute__((ext_vector_type(16))) bf16  v16bf;
typedef __attribute__((ext_vector_type(8)))  bf16  v8bf;
typedef __attribute__((ext_vector_type(8)))  float v8f;

// A fragment: 16(M) x 32(K) bf16 tile from row-major [.., lda].
// ISA layout: lane L -> M = L&15, half = L>>4 ; element e -> K = 16*(e/8) + 8*half + (e%8)
static __device__ inline v16bf load_a_frag(const bf16* base, int lda, int lane) {
  int m = lane & 15, h = lane >> 4;
  const v8bf* p0 = reinterpret_cast<const v8bf*>(base + (long long)m * lda + h * 8);
  const v8bf* p1 = reinterpret_cast<const v8bf*>(base + (long long)m * lda + 16 + h * 8);
  v8bf lo = *p0, hi = *p1;
  v16bf r;
#pragma unroll
  for (int i = 0; i < 8; ++i) { r[i] = lo[i]; r[i + 8] = hi[i]; }
  return r;
}

// B fragment: 32(K) x 16(N) tile of W^T where W is row-major [N, ldw] (weights stored [out,in]).
// ISA layout: lane L -> N = L&15, half = L>>4 ; element e -> K = 16*(e/8) + 8*half + (e%8)
static __device__ inline v16bf load_b_frag(const bf16* base, int ldw, int lane) {
  int n = lane & 15, h = lane >> 4;
  const v8bf* p0 = reinterpret_cast<const v8bf*>(base + (long long)n * ldw + h * 8);
  const v8bf* p1 = reinterpret_cast<const v8bf*>(base + (long long)n * ldw + 16 + h * 8);
  v8bf lo = *p0, hi = *p1;
  v16bf r;
#pragma unroll
  for (int i = 0; i < 8; ++i) { r[i] = lo[i]; r[i + 8] = hi[i]; }
  return r;
}

static __device__ inline v8f wmma_bf16(v16bf a, v16bf b, v8f c) {
  return __builtin_amdgcn_wmma_f32_16x16x32_bf16(false, a, false, b, (short)0, c, false, false);
}

static __device__ inline float sigm(float x) { return 1.0f / (1.0f + __expf(-x)); }

// ---------------- generic bf16 WMMA GEMM: out = act(scale*A@W^T + bias [+aux0]) [post] ----------------
// wave tile: 32(M) x 64(N); software-pipelined k-loop: next k-step's 12 fragment loads are issued
// before the current k-step's 8 WMMAs, so WMMA waits cover loads from a full iteration earlier.
// block = 4 waves stacked in M => 128 x 64 per block. All call sites: M%128==0, N%64==0, K%32==0, K>=64.
__global__ __launch_bounds__(128) void gemm_bf16_kernel(
    const bf16* __restrict__ A, long long sA, int lda,
    const bf16* __restrict__ W, long long sW, int ldw,
    const float* __restrict__ bias, float scale,
    int M, int N, int K,
    int act,          // 0 none, 1 silu, 2 sigmoid
    int post,         // 0 none, 1 v*=aux1, 2 v = v*aux1 + (1-v)*aux2
    int preAdd,       // 1: v += aux0 before activation
    const float* __restrict__ aux0, const float* __restrict__ aux1, const float* __restrict__ aux2,
    float* __restrict__ outF, long long sOF, int ldof,
    bf16* __restrict__ outH, long long sOH, int ldoh,
    int vt, int Tvt)  // vt: store outH transposed per batch-of-Tvt-rows: outH[b][col][t]
{
  const int lane = threadIdx.x & 31;
  const int wave = threadIdx.x >> 5;
  const int m0 = (blockIdx.y * 4 + wave) * 32;
  const int n0 = blockIdx.x * 64;
  if (m0 >= M) return;

  const bf16* Ab = A + (long long)blockIdx.z * sA + (long long)m0 * lda;
  const bf16* Wb = W + (long long)blockIdx.z * sW + (long long)n0 * ldw;
  const long long ldaL = lda, ldwL = ldw;

  // prologue: stage k = 0 fragments
  v16bf a0 = load_a_frag(Ab, lda, lane);
  v16bf a1 = load_a_frag(Ab + 16 * ldaL, lda, lane);
  v16bf b0 = load_b_frag(Wb, ldw, lane);
  v16bf b1 = load_b_frag(Wb + 16 * ldwL, ldw, lane);
  v16bf b2 = load_b_frag(Wb + 32 * ldwL, ldw, lane);
  v16bf b3 = load_b_frag(Wb + 48 * ldwL, ldw, lane);

  v8f acc[8] = {};
  for (int k = 32; k < K; k += 32) {
    // prefetch two k-steps ahead (speculative; dropped if beyond mapping)
    __builtin_prefetch(Ab + k + 32, 0, 1);
    __builtin_prefetch(Wb + k + 32, 0, 1);
    // issue next k-step's loads first ...
    v16bf na0 = load_a_frag(Ab + k, lda, lane);
    v16bf na1 = load_a_frag(Ab + 16 * ldaL + k, lda, lane);
    v16bf nb0 = load_b_frag(Wb + k, ldw, lane);
    v16bf nb1 = load_b_frag(Wb + 16 * ldwL + k, ldw, lane);
    v16bf nb2 = load_b_frag(Wb + 32 * ldwL + k, ldw, lane);
    v16bf nb3 = load_b_frag(Wb + 48 * ldwL + k, ldw, lane);
    // ... then consume the previous k-step's fragments
    acc[0] = wmma_bf16(a0, b0, acc[0]);
    acc[4] = wmma_bf16(a1, b0, acc[4]);
    acc[1] = wmma_bf16(a0, b1, acc[1]);
    acc[5] = wmma_bf16(a1, b1, acc[5]);
    acc[2] = wmma_bf16(a0, b2, acc[2]);
    acc[6] = wmma_bf16(a1, b2, acc[6]);
    acc[3] = wmma_bf16(a0, b3, acc[3]);
    acc[7] = wmma_bf16(a1, b3, acc[7]);
    a0 = na0; a1 = na1; b0 = nb0; b1 = nb1; b2 = nb2; b3 = nb3;
  }
  // epilogue k-step
  acc[0] = wmma_bf16(a0, b0, acc[0]);
  acc[4] = wmma_bf16(a1, b0, acc[4]);
  acc[1] = wmma_bf16(a0, b1, acc[1]);
  acc[5] = wmma_bf16(a1, b1, acc[5]);
  acc[2] = wmma_bf16(a0, b2, acc[2]);
  acc[6] = wmma_bf16(a1, b2, acc[6]);
  acc[3] = wmma_bf16(a0, b3, acc[3]);
  acc[7] = wmma_bf16(a1, b3, acc[7]);

  const int nc = lane & 15, halfm = lane >> 4;
#pragma unroll
  for (int mi = 0; mi < 2; ++mi) {
#pragma unroll
    for (int j = 0; j < 4; ++j) {
      const int col = n0 + j * 16 + nc;
      const float bv = bias ? bias[col] : 0.0f;
#pragma unroll
      for (int r = 0; r < 8; ++r) {
        const int row = m0 + mi * 16 + r + halfm * 8;
        const long long gi = (long long)row * (long long)N + col;  // aux index (non-batched only)
        float v = acc[mi * 4 + j][r] * scale + bv;
        if (preAdd) v += aux0[gi];
        if (act == 1) v = v * sigm(v);
        else if (act == 2) v = sigm(v);
        if (post == 1) v *= aux1[gi];
        else if (post == 2) { const float zc = aux1[gi], bb = aux2[gi]; v = v * zc + (1.0f - v) * bb; }
        if (outF) outF[(long long)blockIdx.z * sOF + (long long)row * ldof + col] = v;
        if (outH) {
          if (vt) {
            const int bi = row / Tvt, t = row % Tvt;
            outH[(long long)bi * sOH + (long long)col * ldoh + t] = (bf16)v;
          } else {
            outH[(long long)blockIdx.z * sOH + (long long)row * ldoh + col] = (bf16)v;
          }
        }
      }
    }
  }
}

// ---------------- EMA recurrence: persistent cooperative kernel ----------------
// 32 blocks x 1 wave; block j owns output columns [16j, 16j+16). Per step:
//   alpha = tanh(prev @ Wap^T + Ua[t]) ; delta = tanh(prev @ Wdp^T + Ud[t])
//   ema   = sigmoid(alpha*b_t + (1 - alpha*delta)*prev)
// prev double-buffered in global (L2-resident); one device-wide barrier per step.
__global__ __launch_bounds__(32) void ema_scan_kernel(
    const bf16* __restrict__ Wap, const bf16* __restrict__ Wdp,   // [512,512] bf16 [out,in]
    const bf16* __restrict__ Ua,  const bf16* __restrict__ Ud,    // [16,1024,512] bf16 (bias incl.)
    const float* __restrict__ bIn,                                // [16,1024,512] f32
    float* __restrict__ prevF, bf16* __restrict__ prevH,          // [2][16*512]
    bf16* __restrict__ emaOut,                                    // [16,1024,512] bf16
    unsigned* __restrict__ barCnt, unsigned* __restrict__ barGen,
    int T, int nb)
{
  const int lane = threadIdx.x & 31;
  const int n0 = blockIdx.x * 16;
  const int nc = lane & 15, halfm = lane >> 4;
  const bf16* WapT = Wap + (long long)n0 * 512;
  const bf16* WdpT = Wdp + (long long)n0 * 512;

  for (int t = 0; t < T; ++t) {
    const bf16*  pc = prevH + (long long)(t & 1) * (16 * 512);
    const float* pf = prevF + (long long)(t & 1) * (16 * 512);
    bf16*  pnH = prevH + (long long)((t + 1) & 1) * (16 * 512);
    float* pnF = prevF + (long long)((t + 1) & 1) * (16 * 512);

    // software-pipelined k-loop (K = 512, 16 steps)
    v16bf a  = load_a_frag(pc, 512, lane);
    v16bf bA = load_b_frag(WapT, 512, lane);
    v16bf bD = load_b_frag(WdpT, 512, lane);
    v8f accA = {}, accD = {};
    for (int k = 32; k < 512; k += 32) {
      v16bf na  = load_a_frag(pc + k, 512, lane);
      v16bf nbA = load_b_frag(WapT + k, 512, lane);
      v16bf nbD = load_b_frag(WdpT + k, 512, lane);
      accA = wmma_bf16(a, bA, accA);
      accD = wmma_bf16(a, bD, accD);
      a = na; bA = nbA; bD = nbD;
    }
    accA = wmma_bf16(a, bA, accA);
    accD = wmma_bf16(a, bD, accD);

#pragma unroll
    for (int r = 0; r < 8; ++r) {
      const int m = r + halfm * 8;              // batch index (M = 16)
      const int n = n0 + nc;
      const long long gi = ((long long)m * T + t) * 512 + n;
      const float alpha = tanhf(accA[r] + (float)Ua[gi]);
      const float delta = tanhf(accD[r] + (float)Ud[gi]);
      const float pv = pf[m * 512 + n];
      const float e = sigm(alpha * bIn[gi] + (1.0f - alpha * delta) * pv);
      emaOut[gi] = (bf16)e;
      pnF[m * 512 + n] = e;
      pnH[m * 512 + n] = (bf16)e;
    }

    // device-wide release/acquire barrier (generation counter; no reset races)
    unsigned g = __hip_atomic_load(barGen, __ATOMIC_RELAXED, __HIP_MEMORY_SCOPE_AGENT);
    __threadfence();
    if (threadIdx.x == 0) {
      unsigned arrived = __hip_atomic_fetch_add(barCnt, 1u, __ATOMIC_ACQ_REL, __HIP_MEMORY_SCOPE_AGENT);
      if (arrived == (unsigned)(nb - 1)) {
        __hip_atomic_store(barCnt, 0u, __ATOMIC_RELAXED, __HIP_MEMORY_SCOPE_AGENT);
        __hip_atomic_fetch_add(barGen, 1u, __ATOMIC_RELEASE, __HIP_MEMORY_SCOPE_AGENT);
      }
    }
    while (__hip_atomic_load(barGen, __ATOMIC_ACQUIRE, __HIP_MEMORY_SCOPE_AGENT) == g)
      __builtin_amdgcn_s_sleep(1);
  }
}

// ---------------- row softmax: f32 in -> bf16 out ----------------
__global__ __launch_bounds__(256) void softmax_kernel(const float* __restrict__ S,
                                                      bf16* __restrict__ P, int n) {
  const long long row = blockIdx.x;
  const float* s = S + row * n;
  bf16* p = P + row * n;
  __shared__ float red[8];
  __shared__ float sMax, sSum;
  const int tid = threadIdx.x, lane = tid & 31, w = tid >> 5;

  float mx = -3.4e38f;
  for (int i = tid; i < n; i += 256) mx = fmaxf(mx, s[i]);
#pragma unroll
  for (int o = 16; o; o >>= 1) mx = fmaxf(mx, __shfl_xor(mx, o, 32));
  if (!lane) red[w] = mx;
  __syncthreads();
  if (!tid) { float m = red[0]; for (int i = 1; i < 8; ++i) m = fmaxf(m, red[i]); sMax = m; }
  __syncthreads();
  mx = sMax;

  float sum = 0.0f;
  for (int i = tid; i < n; i += 256) sum += __expf(s[i] - mx);
#pragma unroll
  for (int o = 16; o; o >>= 1) sum += __shfl_xor(sum, o, 32);
  if (!lane) red[w] = sum;
  __syncthreads();
  if (!tid) { float tt = 0.0f; for (int i = 0; i < 8; ++i) tt += red[i]; sSum = tt; }
  __syncthreads();
  const float inv = 1.0f / sSum;
  for (int i = tid; i < n; i += 256) p[i] = (bf16)(__expf(s[i] - mx) * inv);
}

// ---------------- small utility kernels ----------------
__global__ void cvt_kernel(const float* __restrict__ s, int sld, int soff,
                           bf16* __restrict__ d, long long rows, long long cols) {
  const long long n = rows * cols;
  for (long long i = (long long)blockIdx.x * blockDim.x + threadIdx.x; i < n;
       i += (long long)gridDim.x * blockDim.x) {
    const long long r = i / cols, c = i % cols;
    d[i] = (bf16)s[r * sld + soff + c];
  }
}
__global__ void cvtT_kernel(const float* __restrict__ s, bf16* __restrict__ d, int R, int C) {
  const long long n = (long long)R * C;
  for (long long i = (long long)blockIdx.x * blockDim.x + threadIdx.x; i < n;
       i += (long long)gridDim.x * blockDim.x) {
    const long long r = i / C, c = i % C;
    d[c * (long long)R + r] = (bf16)s[i];
  }
}
__global__ void zero_kernel(unsigned* __restrict__ p, long long n) {
  for (long long i = (long long)blockIdx.x * blockDim.x + threadIdx.x; i < n;
       i += (long long)gridDim.x * blockDim.x) p[i] = 0u;
}

// ---------------- host orchestration ----------------
extern "C" void kernel_launch(void* const* d_in, const int* in_sizes, int n_in,
                              void* d_out, int out_size, void* d_ws, size_t ws_size,
                              hipStream_t stream) {
  (void)in_sizes; (void)n_in; (void)out_size; (void)ws_size;
  constexpr int Bz = 16, T = 1024, D = 512;
  constexpr long long MT = (long long)Bz * T;  // 16384 rows
  constexpr size_t MB = 1ull << 20;

  const float* bIn  = (const float*)d_in[0];
  const float* Wq   = (const float*)d_in[1];  const float* bq   = (const float*)d_in[2];
  const float* Wk   = (const float*)d_in[3];  const float* bk   = (const float*)d_in[4];
  const float* Wv   = (const float*)d_in[5];  const float* bv   = (const float*)d_in[6];
  const float* Wa   = (const float*)d_in[7];  const float* ba   = (const float*)d_in[8];
  const float* Wd   = (const float*)d_in[9];  const float* bd   = (const float*)d_in[10];
  const float* Wout = (const float*)d_in[11]; const float* bout = (const float*)d_in[12];
  const float* W_f  = (const float*)d_in[13]; const float* b_f  = (const float*)d_in[14];
  const float* W_EMA_c = (const float*)d_in[15];
  const float* W_z_C = (const float*)d_in[16]; const float* b_C = (const float*)d_in[17];
  const float* W_i  = (const float*)d_in[18]; const float* b_i  = (const float*)d_in[19];
  const float* W_o  = (const float*)d_in[20]; const float* b_o  = (const float*)d_in[21];
  float* out = (float*)d_out;

  // workspace arena (~215 MB, with reuse)
  char* ws = (char*)d_ws;
  bf16* bbf   = (bf16*)(ws + 0 * MB);     // b in bf16            (dead after Ua/Ud)
  bf16* Ua    = (bf16*)(ws + 16 * MB);    //                      (dead after scan)
  bf16* Ud    = (bf16*)(ws + 32 * MB);    //                      (dead after scan)
  bf16* ema   = (bf16*)(ws + 48 * MB);    //                      (dead after b_ema)
  float* bemaF = (float*)(ws + 64 * MB);  // 32 MB, live throughout
  bf16* bemaH = (bf16*)(ws + 96 * MB);    // live throughout
  bf16* qbuf  = (bf16*)(ws + 0 * MB);     // reuse bbf
  bf16* kbuf  = (bf16*)(ws + 16 * MB);    // reuse Ua
  bf16* vT    = (bf16*)(ws + 32 * MB);    // reuse Ud ; layout [16][D][T]
  float* scores = (float*)(ws + 112 * MB);// 64 MB               (dead after softmax)
  bf16* attn  = (bf16*)(ws + 176 * MB);   // 32 MB               (dead after Z)
  float* Z    = (float*)(ws + 112 * MB);  // reuse scores lower half
  bf16* ZEf   = (bf16*)(ws + 0 * MB);     // reuse qbuf
  float* Gm   = (float*)(ws + 144 * MB);  // reuse scores upper half
  float* ZC   = (float*)(ws + 176 * MB);  // reuse attn
  bf16* Bh    = (bf16*)(ws + 16 * MB);    // reuse kbuf
  // bf16 weights: 13 slots x 512 KB
  char* wbase = ws + 208 * MB;
  auto wslot = [&](int i) { return (bf16*)(wbase + (size_t)i * (512 * 1024)); };
  bf16 *wQ = wslot(0), *wK = wslot(1), *wV = wslot(2), *wOut = wslot(3), *wF = wslot(4),
       *wI = wslot(5), *wO = wslot(6), *wAp = wslot(7), *wAb = wslot(8), *wDp = wslot(9),
       *wDb = wslot(10), *wEmaT = wslot(11), *wZcT = wslot(12);
  // scan state + barrier
  char* sbase = ws + 215 * MB;
  float* prevF = (float*)sbase;                        // 2 x 16*512 f32 (64 KB)
  bf16* prevH = (bf16*)(sbase + 64 * 1024);            // 2 x 16*512 bf16 (32 KB)
  unsigned* barCnt = (unsigned*)(sbase + 96 * 1024);
  unsigned* barGen = barCnt + 1;

  // ---- init: zero scan state + barrier counters (every call: determinism) ----
  zero_kernel<<<64, 256, 0, stream>>>((unsigned*)sbase, (96 * 1024 + 8) / 4);

  // ---- bf16 conversions ----
  cvt_kernel<<<2048, 256, 0, stream>>>(bIn, D, 0, bbf, MT, D);
  cvt_kernel<<<512, 256, 0, stream>>>(Wq, D, 0, wQ, D, D);
  cvt_kernel<<<512, 256, 0, stream>>>(Wk, D, 0, wK, D, D);
  cvt_kernel<<<512, 256, 0, stream>>>(Wv, D, 0, wV, D, D);
  cvt_kernel<<<512, 256, 0, stream>>>(Wout, D, 0, wOut, D, D);
  cvt_kernel<<<512, 256, 0, stream>>>(W_f, D, 0, wF, D, D);
  cvt_kernel<<<512, 256, 0, stream>>>(W_i, D, 0, wI, D, D);
  cvt_kernel<<<512, 256, 0, stream>>>(W_o, D, 0, wO, D, D);
  cvt_kernel<<<512, 256, 0, stream>>>(Wa, 2 * D, 0, wAp, D, D);   // prev half
  cvt_kernel<<<512, 256, 0, stream>>>(Wa, 2 * D, D, wAb, D, D);   // b_t half
  cvt_kernel<<<512, 256, 0, stream>>>(Wd, 2 * D, 0, wDp, D, D);
  cvt_kernel<<<512, 256, 0, stream>>>(Wd, 2 * D, D, wDb, D, D);
  cvtT_kernel<<<512, 256, 0, stream>>>(W_EMA_c, wEmaT, D, D);     // used without .T in ref
  cvtT_kernel<<<512, 256, 0, stream>>>(W_z_C, wZcT, D, D);        // used without .T in ref

  auto gemm = [&](const bf16* A, long long sA, int lda, const bf16* W, long long sW, int ldw,
                  const float* bias, float scale, int M, int N, int K,
                  int act, int post, int preAdd,
                  const float* a0, const float* a1, const float* a2,
                  float* oF, long long sOF, int ldof, bf16* oH, long long sOH, int ldoh,
                  int vt, int batch) {
    dim3 g(N / 64, (M + 127) / 128, batch), blk(128);
    gemm_bf16_kernel<<<g, blk, 0, stream>>>(A, sA, lda, W, sW, ldw, bias, scale, M, N, K,
                                            act, post, preAdd, a0, a1, a2,
                                            oF, sOF, ldof, oH, sOH, ldoh, vt, T);
  };

  // ---- time-parallel halves of the EMA gates: Ua/Ud = b @ Wab^T + bias ----
  gemm(bbf, 0, D, wAb, 0, D, ba, 1.0f, (int)MT, D, D, 0, 0, 0, nullptr, nullptr, nullptr,
       nullptr, 0, 0, Ua, 0, D, 0, 1);
  gemm(bbf, 0, D, wDb, 0, D, bd, 1.0f, (int)MT, D, D, 0, 0, 0, nullptr, nullptr, nullptr,
       nullptr, 0, 0, Ud, 0, D, 0, 1);

  // ---- sequential EMA scan (persistent, 32 single-wave blocks) ----
  ema_scan_kernel<<<32, 32, 0, stream>>>(wAp, wDp, Ua, Ud, bIn, prevF, prevH, ema,
                                         barCnt, barGen, T, 32);

  // ---- b_ema = silu(ema @ Wout^T + bout), keep f32 + bf16 ----
  gemm(ema, 0, D, wOut, 0, D, bout, 1.0f, (int)MT, D, D, 1, 0, 0, nullptr, nullptr, nullptr,
       bemaF, 0, D, bemaH, 0, D, 0, 1);

  // ---- q, k, v projections (v stored transposed per batch: vT[b][d][t]) ----
  gemm(bemaH, 0, D, wQ, 0, D, bq, 1.0f, (int)MT, D, D, 0, 0, 0, nullptr, nullptr, nullptr,
       nullptr, 0, 0, qbuf, 0, D, 0, 1);
  gemm(bemaH, 0, D, wK, 0, D, bk, 1.0f, (int)MT, D, D, 0, 0, 0, nullptr, nullptr, nullptr,
       nullptr, 0, 0, kbuf, 0, D, 0, 1);
  gemm(bemaH, 0, D, wV, 0, D, bv, 1.0f, (int)MT, D, D, 0, 0, 0, nullptr, nullptr, nullptr,
       nullptr, 0, 0, vT, (long long)D * T, T, 1, 1);

  // ---- scores = q @ k^T / sqrt(D)  (batched over 16) ----
  gemm(qbuf, (long long)T * D, D, kbuf, (long long)T * D, D, nullptr,
       1.0f / sqrtf((float)D), T, T, D, 0, 0, 0, nullptr, nullptr, nullptr,
       scores, (long long)T * T, T, nullptr, 0, 0, 0, Bz);

  // ---- softmax rows -> attn (bf16) ----
  softmax_kernel<<<(int)MT, 256, 0, stream>>>(scores, attn, T);

  // ---- Z_EMA = attn @ v  (== attn @ (vT)^T, batched) ----
  gemm(attn, (long long)T * T, T, vT, (long long)D * T, T, nullptr, 1.0f,
       T, D, T, 0, 0, 0, nullptr, nullptr, nullptr,
       Z, (long long)T * D, D, nullptr, 0, 0, 0, Bz);

  // ---- ZEf = sigmoid(b_ema @ W_f^T + b_f) * Z ----
  gemm(bemaH, 0, D, wF, 0, D, b_f, 1.0f, (int)MT, D, D, 2, 1, 0, nullptr, Z, nullptr,
       nullptr, 0, 0, ZEf, 0, D, 0, 1);

  // ---- Gm = b_ema @ W_EMA_c ----
  gemm(bemaH, 0, D, wEmaT, 0, D, nullptr, 1.0f, (int)MT, D, D, 0, 0, 0, nullptr, nullptr,
       nullptr, Gm, 0, D, nullptr, 0, 0, 0, 1);

  // ---- ZC = silu(Gm + ZEf @ W_z_C + b_C) ----
  gemm(ZEf, 0, D, wZcT, 0, D, b_C, 1.0f, (int)MT, D, D, 1, 0, 1, Gm, nullptr, nullptr,
       ZC, 0, D, nullptr, 0, 0, 0, 1);

  // ---- Bh = i*ZC + (1-i)*b,  i = sigmoid(b_ema @ W_i^T + b_i) ----
  gemm(bemaH, 0, D, wI, 0, D, b_i, 1.0f, (int)MT, D, D, 2, 2, 0, nullptr, ZC, bIn,
       nullptr, 0, 0, Bh, 0, D, 0, 1);

  // ---- out = sigmoid(Bh @ W_o^T + b_o) ----
  gemm(Bh, 0, D, wO, 0, D, b_o, 1.0f, (int)MT, D, D, 2, 0, 0, nullptr, nullptr, nullptr,
       out, 0, D, nullptr, 0, 0, 0, 1);
}
